// AutoEncoder_4836133175590
// MI455X (gfx1250) — matrix operand order repaired and checked by reading, not compiled, not run on hardware
//
#include <hip/hip_runtime.h>
#include <hip/hip_bf16.h>

// ---------------------------------------------------------------------------
// Problem constants
// ---------------------------------------------------------------------------
#define NTOT  262144
#define BSETS 16384
#define SET   16
#define DIM   128
#define HID   128
#define MN    17

typedef __attribute__((ext_vector_type(16))) _Float16 v16h;
typedef __attribute__((ext_vector_type(8)))  _Float16 v8h;
typedef __attribute__((ext_vector_type(8)))  float    v8f;

// Packed-weight offsets (units: halves). frag = 512 halves (32 lanes x 16).
constexpr size_t PW_VDS0 = 0;        // 128x128  KF4 NT8
constexpr size_t PW_VDS1 = 16384;    // 128x128
constexpr size_t PW_EDS0 = 32768;    // 128x128
constexpr size_t PW_EDS1 = 49152;    // 128x128
constexpr size_t PW_VM0  = 65536;    // 256x192  KF8 NT12
constexpr size_t PW_VM1  = 114688;   // 192x128  KF6 NT8
constexpr size_t PW_KM0  = 139264;   // 145x136 -> 160x144  KF5 NT9
constexpr size_t PW_KM1  = 162304;   // 136x128 -> 160x128  KF5 NT8
constexpr size_t PW_EM0  = 182784;   // 145x136 -> 160x144
constexpr size_t PW_EM1  = 205824;   // 136x128 -> 160x128
constexpr size_t PW_SP0  = 226304;   // 128x72  -> 128x80   KF4 NT5
constexpr size_t PW_SP1  = 236544;   // 72x17   -> 96x32    KF3 NT2
constexpr size_t PW_DEC0 = 239616;   // 128x128
constexpr size_t PW_DEC1 = 256000;   // 128x128
constexpr size_t PW_TOTAL= 272384;

// ---------------------------------------------------------------------------
// WMMA fragment loads
// A: 16xK row-major (global or LDS). 16-bit A layout: lane<16 holds row=lane,
//    halves0-7 = K (kb*32 + 0..7), halves8-15 = K (kb*32 + 16..23);
//    lane>=16 same row(lane-16) with +8 K offset.
// B: pre-packed so each lane reads its 16 halves contiguously.
// ---------------------------------------------------------------------------
__device__ __forceinline__ v16h ld_a(const _Float16* A, int lda, int kb, int lane) {
  const int m = lane & 15, hi = lane >> 4;
  const _Float16* p = A + (size_t)m * lda + kb * 32 + hi * 8;
  const v8h c0 = *(const v8h*)(p);
  const v8h c1 = *(const v8h*)(p + 16);
  v16h a;
#pragma unroll
  for (int i = 0; i < 8; ++i) { a[i] = c0[i]; a[i + 8] = c1[i]; }
  return a;
}

__device__ __forceinline__ v16h ld_b(const _Float16* Bp, int frag, int lane) {
  const _Float16* p = Bp + ((size_t)frag << 9) + (lane << 4);
  const v8h c0 = *(const v8h*)(p);
  const v8h c1 = *(const v8h*)(p + 8);
  v16h b;
#pragma unroll
  for (int i = 0; i < 8; ++i) { b[i] = c0[i]; b[i + 8] = c1[i]; }
  return b;
}

// One-wave GEMM: C[16 x NT*16] = A[16 x KF*32] * B, f32 accumulate.
template<int NT, int KF>
__device__ __forceinline__ void wave_gemm(v8f* acc, const _Float16* A, int lda,
                                          const _Float16* Bp, int lane) {
#pragma unroll
  for (int nb = 0; nb < NT; ++nb)
#pragma unroll
    for (int j = 0; j < 8; ++j) acc[nb][j] = 0.f;
#pragma unroll
  for (int kb = 0; kb < KF; ++kb) {
    v16h a = ld_a(A, lda, kb, lane);
#pragma unroll
    for (int nb = 0; nb < NT; ++nb) {
      v16h b = ld_b(Bp, kb * NT + nb, lane);
      acc[nb] = __builtin_amdgcn_wmma_f32_16x16x32_f16(
          false, a, false, b, (short)0, acc[nb], false, false);
    }
  }
}

// ---------------------------------------------------------------------------
// Epilogues on C layout: lane holds col = nb*16 + (lane&15),
// VGPR j holds row = j + 8*(lane>>4).
// ---------------------------------------------------------------------------
template<int NT>
__device__ __forceinline__ void add_bias(v8f* acc, const float* b, int F, int lane) {
  const int c0 = lane & 15;
#pragma unroll
  for (int nb = 0; nb < NT; ++nb) {
    const int col = nb * 16 + c0;
    const float bb = (col < F) ? b[col] : 0.f;
#pragma unroll
    for (int j = 0; j < 8; ++j) acc[nb][j] += bb;
  }
}

// LayerNorm(feature dim F, masked) * g + beta, tanh, store f16 rows to LDS.
template<int NT>
__device__ __forceinline__ void ln_tanh_to_lds(v8f* acc, int F, const float* g,
                                               const float* bt, _Float16* out,
                                               int ldo, int lane) {
  const int c0 = lane & 15, hi = lane >> 4;
  float sm[8], sq[8];
#pragma unroll
  for (int j = 0; j < 8; ++j) { sm[j] = 0.f; sq[j] = 0.f; }
#pragma unroll
  for (int nb = 0; nb < NT; ++nb) {
    const int col = nb * 16 + c0;
    if (col < F) {
#pragma unroll
      for (int j = 0; j < 8; ++j) { float x = acc[nb][j]; sm[j] += x; sq[j] += x * x; }
    }
  }
#pragma unroll
  for (int j = 0; j < 8; ++j)
    for (int m = 1; m < 16; m <<= 1) {
      sm[j] += __shfl_xor(sm[j], m, 32);
      sq[j] += __shfl_xor(sq[j], m, 32);
    }
  const float invF = 1.f / (float)F;
  float mean[8], rstd[8];
#pragma unroll
  for (int j = 0; j < 8; ++j) {
    mean[j] = sm[j] * invF;
    float var = sq[j] * invF - mean[j] * mean[j];
    rstd[j] = rsqrtf(var + 1e-5f);
  }
#pragma unroll
  for (int nb = 0; nb < NT; ++nb) {
    const int col = nb * 16 + c0;
    const float gg = (col < F) ? g[col] : 0.f;
    const float bb = (col < F) ? bt[col] : 0.f;
#pragma unroll
    for (int j = 0; j < 8; ++j) {
      float t = tanhf((acc[nb][j] - mean[j]) * rstd[j] * gg + bb);
      out[(size_t)(j + 8 * hi) * ldo + col] = (_Float16)t;
    }
  }
}

template<int NT>
__device__ __forceinline__ void tanh_to_lds(v8f* acc, _Float16* out, int ldo, int lane) {
  const int c0 = lane & 15, hi = lane >> 4;
#pragma unroll
  for (int nb = 0; nb < NT; ++nb)
#pragma unroll
    for (int j = 0; j < 8; ++j)
      out[(size_t)(j + 8 * hi) * ldo + nb * 16 + c0] = (_Float16)tanhf(acc[nb][j]);
}

__device__ __forceinline__ void zero_cols(_Float16* out, int ldo, int cfrom, int cto, int lane) {
  const int w = cto - cfrom;
  for (int idx = lane; idx < 16 * w; idx += 32) {
    int t = idx / w, c = cfrom + (idx - t * w);
    out[(size_t)t * ldo + c] = (_Float16)0.f;
  }
}

// Segment-sum over 16 rows -> f16 row of NT*16 features.
template<int NT>
__device__ __forceinline__ void colsum_store(v8f* acc, _Float16* out, int lane) {
#pragma unroll
  for (int nb = 0; nb < NT; ++nb) {
    float s = 0.f;
#pragma unroll
    for (int j = 0; j < 8; ++j) s += acc[nb][j];
    s += __shfl_xor(s, 16, 32);
    if (lane < 16) out[nb * 16 + lane] = (_Float16)s;
  }
}

template<int NT>
__device__ __forceinline__ void store_c_h(v8f* acc, _Float16* out, int ldo, int lane) {
  const int c0 = lane & 15, hi = lane >> 4;
#pragma unroll
  for (int nb = 0; nb < NT; ++nb)
#pragma unroll
    for (int j = 0; j < 8; ++j)
      out[(size_t)(j + 8 * hi) * ldo + nb * 16 + c0] = (_Float16)acc[nb][j];
}

// ---------------------------------------------------------------------------
// Weight pack: row-major f32 [K,N] -> wave32 B-fragment layout, f16, zero pad.
// ---------------------------------------------------------------------------
__global__ void k_pack(const float* __restrict__ W, _Float16* __restrict__ dst,
                       int K, int Ncols, int KF, int NT) {
  const int idx = blockIdx.x * 256 + threadIdx.x;
  const int total = KF * NT * 512;
  if (idx >= total) return;
  const int frag = idx >> 9, rem = idx & 511;
  const int lane = rem >> 4, j = rem & 15;
  const int kb = frag / NT, nb = frag - kb * NT;
  const int k = kb * 32 + (lane >> 4) * 8 + (j & 7) + (j >> 3) * 16;
  const int n = nb * 16 + (lane & 15);
  const float v = (k < K && n < Ncols) ? W[(size_t)k * Ncols + n] : 0.f;
  dst[idx] = (_Float16)v;
}

// ---------------------------------------------------------------------------
// Tiny tables: key_ds(one-hot pos 0..15) -> f32 [16,128];
//              key_dec(eye(17))          -> f16 [17,128].
// Both are MLP 17 -> 72 (LN, tanh) -> 128 with a one-hot input row.
// ---------------------------------------------------------------------------
__global__ void k_prep(const float* dsW0, const float* dsW1, const float* dsb0,
                       const float* dsb1, const float* dsg, const float* dsbt,
                       const float* dcW0, const float* dcW1, const float* dcb0,
                       const float* dcb1, const float* dcg, const float* dcbt,
                       float* ktbl, _Float16* keys) {
  const int t = threadIdx.x;
  if (t >= 33) return;
  const bool isds = (t < 16);
  const int r = isds ? t : (t - 16);
  const float* W0 = isds ? dsW0 : dcW0;
  const float* W1 = isds ? dsW1 : dcW1;
  const float* b0 = isds ? dsb0 : dcb0;
  const float* b1 = isds ? dsb1 : dcb1;
  const float* g  = isds ? dsg  : dcg;
  const float* bt = isds ? dsbt : dcbt;
  float h[72];
#pragma unroll
  for (int j = 0; j < 72; ++j) h[j] = W0[r * 72 + j] + b0[j];
  float m = 0.f;
#pragma unroll
  for (int j = 0; j < 72; ++j) m += h[j];
  m *= (1.f / 72.f);
  float v = 0.f;
#pragma unroll
  for (int j = 0; j < 72; ++j) { float d = h[j] - m; v += d * d; }
  v *= (1.f / 72.f);
  const float rs = rsqrtf(v + 1e-5f);
#pragma unroll
  for (int j = 0; j < 72; ++j) h[j] = tanhf((h[j] - m) * rs * g[j] + bt[j]);
  for (int c = 0; c < 128; ++c) {
    float s = b1[c];
#pragma unroll
    for (int j = 0; j < 72; ++j) s += h[j] * W1[j * 128 + c];
    if (isds) ktbl[r * 128 + c] = s;
    else      keys[r * 128 + c] = (_Float16)s;
  }
}

// ---------------------------------------------------------------------------
// Rank, stable per-set sort (16 elems), gather + f32->f16 convert.
// One wave per set.
// ---------------------------------------------------------------------------
__global__ void k_sort(const float* __restrict__ x, const float* __restrict__ rw,
                       const float* __restrict__ rb, _Float16* __restrict__ xs) {
  __shared__ float mag[16];
  __shared__ int   ord[16];
  const int s = blockIdx.x, lane = threadIdx.x;
  const int r = lane & 15, hf = lane >> 4;
  const float* xp = x + ((size_t)s * SET + r) * DIM + hf * 64;
  float d = 0.f;
#pragma unroll
  for (int i = 0; i < 64; ++i) d += xp[i] * rw[hf * 64 + i];
  d += __shfl_xor(d, 16, 32);
  if (lane < 16) mag[lane] = d + rb[0];
  __syncthreads();
  if (lane < 16) {
    const float mi = mag[lane];
    int rank = 0;
#pragma unroll
    for (int j = 0; j < 16; ++j) {
      const float mj = mag[j];
      rank += (mj < mi) || (mj == mi && j < lane);
    }
    ord[rank] = lane;
  }
  __syncthreads();
  for (int idx = lane; idx < SET * DIM; idx += 32) {
    const int p = idx >> 7, c = idx & 127;
    xs[(size_t)s * (SET * DIM) + idx] =
        (_Float16)x[((size_t)s * SET + ord[p]) * DIM + c];
  }
}

// ---------------------------------------------------------------------------
// Deepset branch per set: y2_ds[s] = sum_rows( val_ds(xs) * key_ds_table[pos] )
// ---------------------------------------------------------------------------
__global__ void k_e1(const _Float16* __restrict__ xs, const _Float16* __restrict__ pw,
                     const float* b0, const float* g, const float* bt, const float* b1,
                     const float* __restrict__ ktbl, _Float16* __restrict__ y2ds) {
  __shared__ alignas(16) _Float16 U[16 * 128];
  const int s = blockIdx.x, lane = threadIdx.x;
  const _Float16* A = xs + (size_t)s * (SET * DIM);
  v8f acc[8];
  wave_gemm<8, 4>(acc, A, 128, pw + PW_VDS0, lane);
  add_bias<8>(acc, b0, 128, lane);
  ln_tanh_to_lds<8>(acc, 128, g, bt, U, 128, lane);
  __syncthreads();
  wave_gemm<8, 4>(acc, U, 128, pw + PW_VDS1, lane);
  add_bias<8>(acc, b1, 128, lane);
  const int c0 = lane & 15, hi = lane >> 4;
#pragma unroll
  for (int nb = 0; nb < 8; ++nb)
#pragma unroll
    for (int j = 0; j < 8; ++j)
      acc[nb][j] *= ktbl[(j + 8 * hi) * 128 + nb * 16 + c0];
  colsum_store<8>(acc, y2ds + (size_t)s * 128, lane);
}

// z_ds = enc_ds(y2_ds): 16 sets per wave (rows = sets), tanh only.
__global__ void k_s1(const _Float16* __restrict__ y2ds, const _Float16* __restrict__ pw,
                     const float* b0, const float* b1, _Float16* __restrict__ zds) {
  __shared__ alignas(16) _Float16 U[16 * 128];
  const int blk = blockIdx.x, lane = threadIdx.x;
  const _Float16* A = y2ds + (size_t)blk * 16 * 128;
  v8f acc[8];
  wave_gemm<8, 4>(acc, A, 128, pw + PW_EDS0, lane);
  add_bias<8>(acc, b0, 128, lane);
  tanh_to_lds<8>(acc, U, 128, lane);
  __syncthreads();
  wave_gemm<8, 4>(acc, U, 128, pw + PW_EDS1, lane);
  add_bias<8>(acc, b1, 128, lane);
  store_c_h<8>(acc, zds + (size_t)blk * 16 * 128, 128, lane);
}

// ---------------------------------------------------------------------------
// Main branch per set: y2[s] = sum_rows( val_main([xs,z]) * key_main([pos,z]) )
// ---------------------------------------------------------------------------
__global__ void k_e2(const _Float16* __restrict__ xs, const _Float16* __restrict__ zds,
                     const _Float16* __restrict__ pw,
                     const float* vm_b0, const float* vm_g, const float* vm_bt, const float* vm_b1,
                     const float* km_b0, const float* km_g, const float* km_bt, const float* km_b1,
                     _Float16* __restrict__ y2m) {
  __shared__ alignas(16) _Float16 bufA[16 * 256];
  __shared__ alignas(16) _Float16 U[16 * 192];
  const int s = blockIdx.x, lane = threadIdx.x;
  // Stage A1 = concat([xs rows, z_ds[s] broadcast])  (16 x 256)
  for (int idx = lane; idx < 16 * 256; idx += 32) {
    const int t = idx >> 8, c = idx & 255;
    bufA[idx] = (c < 128) ? xs[((size_t)s * SET + t) * DIM + c]
                          : zds[(size_t)s * 128 + (c - 128)];
  }
  __syncthreads();
  v8f accV[8];
  {
    v8f acc[12];
    wave_gemm<12, 8>(acc, bufA, 256, pw + PW_VM0, lane);
    add_bias<12>(acc, vm_b0, 192, lane);
    ln_tanh_to_lds<12>(acc, 192, vm_g, vm_bt, U, 192, lane);
  }
  __syncthreads();
  wave_gemm<8, 6>(accV, U, 192, pw + PW_VM1, lane);
  add_bias<8>(accV, vm_b1, 128, lane);
  __syncthreads();
  // Stage A2 = concat([pos one-hot, z_ds[s]]) padded to 160 (16 x 160)
  for (int idx = lane; idx < 16 * 160; idx += 32) {
    const int t = idx / 160, c = idx - t * 160;
    _Float16 v = (_Float16)0.f;
    if (c < 17) { if (c == t) v = (_Float16)1.f; }
    else if (c < 145) v = zds[(size_t)s * 128 + (c - 17)];
    bufA[t * 160 + c] = v;
  }
  __syncthreads();
  {
    v8f acc[9];
    wave_gemm<9, 5>(acc, bufA, 160, pw + PW_KM0, lane);
    add_bias<9>(acc, km_b0, 136, lane);
    ln_tanh_to_lds<9>(acc, 136, km_g, km_bt, U, 160, lane);
    zero_cols(U, 160, 144, 160, lane);
  }
  __syncthreads();
  {
    v8f acc[8];
    wave_gemm<8, 5>(acc, U, 160, pw + PW_KM1, lane);
    add_bias<8>(acc, km_b1, 128, lane);
#pragma unroll
    for (int nb = 0; nb < 8; ++nb)
#pragma unroll
      for (int j = 0; j < 8; ++j) acc[nb][j] *= accV[nb][j];
    colsum_store<8>(acc, y2m + (size_t)s * 128, lane);
  }
}

// z = enc_main(concat([y2, one_hot(16)])), tanh only. 16 sets per wave.
__global__ void k_s2(const _Float16* __restrict__ y2m, const _Float16* __restrict__ pw,
                     const float* b0, const float* b1, _Float16* __restrict__ zh) {
  __shared__ alignas(16) _Float16 bufA[16 * 160];
  __shared__ alignas(16) _Float16 U[16 * 160];
  const int blk = blockIdx.x, lane = threadIdx.x;
  for (int idx = lane; idx < 16 * 160; idx += 32) {
    const int t = idx / 160, c = idx - t * 160;
    _Float16 v = (_Float16)0.f;
    if (c < 128) v = y2m[((size_t)blk * 16 + t) * 128 + c];
    else if (c == 144) v = (_Float16)1.f;   // pos_n = one_hot(n=16)
    bufA[t * 160 + c] = v;
  }
  __syncthreads();
  {
    v8f acc[9];
    wave_gemm<9, 5>(acc, bufA, 160, pw + PW_EM0, lane);
    add_bias<9>(acc, b0, 136, lane);
    tanh_to_lds<9>(acc, U, 160, lane);
    zero_cols(U, 160, 144, 160, lane);
  }
  __syncthreads();
  {
    v8f acc[8];
    wave_gemm<8, 5>(acc, U, 160, pw + PW_EM1, lane);
    add_bias<8>(acc, b1, 128, lane);
    store_c_h<8>(acc, zh + (size_t)blk * 16 * 128, 128, lane);
  }
}

// size_pred + argmax + mask. 16 sets per wave.
__global__ void k_s3(const _Float16* __restrict__ zh, const _Float16* __restrict__ pw,
                     const float* b0, const float* g, const float* bt, const float* b1,
                     int* __restrict__ npred, float* __restrict__ out_mask,
                     float* __restrict__ out_np) {
  __shared__ alignas(16) _Float16 U[16 * 96];
  const int blk = blockIdx.x, lane = threadIdx.x;
  const _Float16* A = zh + (size_t)blk * 16 * 128;
  {
    v8f acc[5];
    wave_gemm<5, 4>(acc, A, 128, pw + PW_SP0, lane);
    add_bias<5>(acc, b0, 72, lane);
    ln_tanh_to_lds<5>(acc, 72, g, bt, U, 96, lane);
    zero_cols(U, 96, 80, 96, lane);
  }
  __syncthreads();
  v8f acc2[2];
  wave_gemm<2, 3>(acc2, U, 96, pw + PW_SP1, lane);
  add_bias<2>(acc2, b1, 17, lane);
  // argmax over 17 logits per row (first-max semantics)
  const int c0 = lane & 15, hi = lane >> 4;
  float bv[8]; int bi[8];
#pragma unroll
  for (int j = 0; j < 8; ++j) { bv[j] = -3.0e38f; bi[j] = 0x7fffffff; }
#pragma unroll
  for (int nb = 0; nb < 2; ++nb) {
    const int col = nb * 16 + c0;
    const bool valid = col < 17;
#pragma unroll
    for (int j = 0; j < 8; ++j) {
      const float v = valid ? acc2[nb][j] : -3.0e38f;
      if (v > bv[j] || (v == bv[j] && col < bi[j])) { bv[j] = v; bi[j] = col; }
    }
  }
#pragma unroll
  for (int j = 0; j < 8; ++j)
    for (int m = 1; m < 16; m <<= 1) {
      const float ov = __shfl_xor(bv[j], m, 32);
      const int   oi = __shfl_xor(bi[j], m, 32);
      if (ov > bv[j] || (ov == bv[j] && oi < bi[j])) { bv[j] = ov; bi[j] = oi; }
    }
  if ((lane & 15) == 0) {
#pragma unroll
    for (int j = 0; j < 8; ++j) {
      const int row = j + 8 * hi;
      const int gs = blk * 16 + row;
      const int np = bi[j];
      npred[gs] = np;
      out_np[gs] = (float)np;
      for (int c = 0; c < MN; ++c)
        out_mask[(size_t)gs * MN + c] = (c < np) ? 1.f : 0.f;
    }
  }
}

// Decoder: 16 rows of flattened (b, j) per wave; A row = z[b]*keys_all[j].
__global__ void k_d(const _Float16* __restrict__ zh, const _Float16* __restrict__ keys,
                    const _Float16* __restrict__ pw, const float* b0, const float* b1,
                    const int* __restrict__ npred, float* __restrict__ xr) {
  __shared__ alignas(16) _Float16 bufA[16 * 128];
  __shared__ alignas(16) _Float16 U[16 * 128];
  const int r0 = blockIdx.x * 16, lane = threadIdx.x;
  for (int idx = lane; idx < 16 * 128; idx += 32) {
    const int t = idx >> 7, c = idx & 127;
    const int r = r0 + t, b = r / MN, jj = r - b * MN;
    bufA[idx] = (_Float16)((float)zh[(size_t)b * 128 + c] *
                           (float)keys[(size_t)jj * 128 + c]);
  }
  __syncthreads();
  v8f acc[8];
  wave_gemm<8, 4>(acc, bufA, 128, pw + PW_DEC0, lane);
  add_bias<8>(acc, b0, 128, lane);
  tanh_to_lds<8>(acc, U, 128, lane);
  __syncthreads();
  wave_gemm<8, 4>(acc, U, 128, pw + PW_DEC1, lane);
  add_bias<8>(acc, b1, 128, lane);
  const int c0 = lane & 15, hi = lane >> 4;
#pragma unroll
  for (int j = 0; j < 8; ++j) {
    const int r = r0 + j + 8 * hi, b = r / MN, jj = r - b * MN;
    const float mk = (jj < npred[b]) ? 1.f : 0.f;
#pragma unroll
    for (int nb = 0; nb < 8; ++nb)
      xr[(size_t)r * 128 + nb * 16 + c0] = acc[nb][j] * mk;
  }
}

// ---------------------------------------------------------------------------
// Host launcher
// ---------------------------------------------------------------------------
extern "C" void kernel_launch(void* const* d_in, const int* in_sizes, int n_in,
                              void* d_out, int out_size, void* d_ws, size_t ws_size,
                              hipStream_t stream) {
  (void)in_sizes; (void)n_in; (void)out_size; (void)ws_size;
  // d_in flat order (setup_inputs dict order; each MLP: W0,W1,b0,b1,g0,beta0):
  const float* x       = (const float*)d_in[0];
  // [1] batch (unused: segments are contiguous size-16)
  const float* kdsW0   = (const float*)d_in[2];
  const float* kdsW1   = (const float*)d_in[3];
  const float* kdsb0   = (const float*)d_in[4];
  const float* kdsb1   = (const float*)d_in[5];
  const float* kdsg    = (const float*)d_in[6];
  const float* kdsbt   = (const float*)d_in[7];
  const float* vdsW0   = (const float*)d_in[8];
  const float* vdsW1   = (const float*)d_in[9];
  const float* vdsb0   = (const float*)d_in[10];
  const float* vdsb1   = (const float*)d_in[11];
  const float* vdsg    = (const float*)d_in[12];
  const float* vdsbt   = (const float*)d_in[13];
  const float* edsW0   = (const float*)d_in[14];
  const float* edsW1   = (const float*)d_in[15];
  const float* edsb0   = (const float*)d_in[16];
  const float* edsb1   = (const float*)d_in[17];
  const float* kmW0    = (const float*)d_in[20];
  const float* kmW1    = (const float*)d_in[21];
  const float* kmb0    = (const float*)d_in[22];
  const float* kmb1    = (const float*)d_in[23];
  const float* kmg     = (const float*)d_in[24];
  const float* kmbt    = (const float*)d_in[25];
  const float* vmW0    = (const float*)d_in[26];
  const float* vmW1    = (const float*)d_in[27];
  const float* vmb0    = (const float*)d_in[28];
  const float* vmb1    = (const float*)d_in[29];
  const float* vmg     = (const float*)d_in[30];
  const float* vmbt    = (const float*)d_in[31];
  const float* emW0    = (const float*)d_in[32];
  const float* emW1    = (const float*)d_in[33];
  const float* emb0    = (const float*)d_in[34];
  const float* emb1    = (const float*)d_in[35];
  const float* rank_w  = (const float*)d_in[38];
  const float* rank_b  = (const float*)d_in[39];
  const float* kdcW0   = (const float*)d_in[40];
  const float* kdcW1   = (const float*)d_in[41];
  const float* kdcb0   = (const float*)d_in[42];
  const float* kdcb1   = (const float*)d_in[43];
  const float* kdcg    = (const float*)d_in[44];
  const float* kdcbt   = (const float*)d_in[45];
  const float* decW0   = (const float*)d_in[46];
  const float* decW1   = (const float*)d_in[47];
  const float* decb0   = (const float*)d_in[48];
  const float* decb1   = (const float*)d_in[49];
  const float* spW0    = (const float*)d_in[52];
  const float* spW1    = (const float*)d_in[53];
  const float* spb0    = (const float*)d_in[54];
  const float* spb1    = (const float*)d_in[55];
  const float* spg     = (const float*)d_in[56];
  const float* spbt    = (const float*)d_in[57];

  // Workspace layout (256B aligned regions)
  char* ws = (char*)d_ws;
  size_t off = 0;
  auto take = [&](size_t bytes) -> char* {
    char* p = ws + off;
    off = (off + bytes + 255) & ~(size_t)255;
    return p;
  };
  _Float16* xs    = (_Float16*)take((size_t)NTOT * DIM * 2);
  float*    ktbl  = (float*)   take((size_t)16 * 128 * 4);
  _Float16* keys  = (_Float16*)take((size_t)MN * 128 * 2);
  _Float16* y2ds  = (_Float16*)take((size_t)BSETS * 128 * 2);
  _Float16* zds   = (_Float16*)take((size_t)BSETS * 128 * 2);
  _Float16* y2m   = (_Float16*)take((size_t)BSETS * 128 * 2);
  _Float16* zh    = (_Float16*)take((size_t)BSETS * 128 * 2);
  int*      npred = (int*)     take((size_t)BSETS * 4);
  _Float16* pw    = (_Float16*)take(PW_TOTAL * 2);

  float* out_xr   = (float*)d_out;
  float* out_mask = out_xr + (size_t)BSETS * MN * DIM;
  float* out_np   = out_mask + (size_t)BSETS * MN;

  // 1) Pack all GEMM weights into wave32 B-fragment layout (f16, zero-padded).
  auto pack = [&](const float* W, size_t dstOff, int K, int Nc, int KF, int NT) {
    const int total = KF * NT * 512;
    k_pack<<<(total + 255) / 256, 256, 0, stream>>>(W, pw + dstOff, K, Nc, KF, NT);
  };
  pack(vdsW0, PW_VDS0, 128, 128, 4, 8);
  pack(vdsW1, PW_VDS1, 128, 128, 4, 8);
  pack(edsW0, PW_EDS0, 128, 128, 4, 8);
  pack(edsW1, PW_EDS1, 128, 128, 4, 8);
  pack(vmW0,  PW_VM0,  256, 192, 8, 12);
  pack(vmW1,  PW_VM1,  192, 128, 6, 8);
  pack(kmW0,  PW_KM0,  145, 136, 5, 9);
  pack(kmW1,  PW_KM1,  136, 128, 5, 8);
  pack(emW0,  PW_EM0,  145, 136, 5, 9);
  pack(emW1,  PW_EM1,  136, 128, 5, 8);
  pack(spW0,  PW_SP0,  128, 72,  4, 5);
  pack(spW1,  PW_SP1,  72,  17,  3, 2);
  pack(decW0, PW_DEC0, 128, 128, 4, 8);
  pack(decW1, PW_DEC1, 128, 128, 4, 8);

  // 2) Tiny tables: key_ds positional table + key_dec keys_all.
  k_prep<<<1, 64, 0, stream>>>(kdsW0, kdsW1, kdsb0, kdsb1, kdsg, kdsbt,
                               kdcW0, kdcW1, kdcb0, kdcb1, kdcg, kdcbt,
                               ktbl, keys);

  // 3) Rank + per-set stable sort + gather to f16.
  k_sort<<<BSETS, 32, 0, stream>>>(x, rank_w, rank_b, xs);

  // 4) Deepset branch (per set, fused 2-layer WMMA + LN + key mult + segsum).
  k_e1<<<BSETS, 32, 0, stream>>>(xs, pw, vdsb0, vdsg, vdsbt, vdsb1, ktbl, y2ds);

  // 5) z_ds = enc_ds(y2_ds)   (16 sets per wave)
  k_s1<<<BSETS / 16, 32, 0, stream>>>(y2ds, pw, edsb0, edsb1, zds);

  // 6) Main branch (per set, fused 4 GEMMs).
  k_e2<<<BSETS, 32, 0, stream>>>(xs, zds, pw,
                                 vmb0, vmg, vmbt, vmb1,
                                 kmb0, kmg, kmbt, kmb1, y2m);

  // 7) z = enc_main([y2, one_hot(16)])
  k_s2<<<BSETS / 16, 32, 0, stream>>>(y2m, pw, emb0, emb1, zh);

  // 8) size_pred + argmax + mask outputs.
  k_s3<<<BSETS / 16, 32, 0, stream>>>(zh, pw, spb0, spg, spbt, spb1,
                                      npred, out_mask, out_np);

  // 9) Decoder over B*17 rows, masked store of xr.
  k_d<<<(BSETS * MN) / 16, 32, 0, stream>>>(zh, keys, pw, decb0, decb1, npred, out_xr);
}
